// Stage1_32865089749702
// MI455X (gfx1250) — matrix operand order repaired and checked by reading, not compiled, not run on hardware
//
#include <hip/hip_runtime.h>

#define BN_EPS_F 1e-5f

typedef __attribute__((ext_vector_type(16))) __bf16 v16bf;
typedef __attribute__((ext_vector_type(8)))  __bf16 v8bf;
typedef __attribute__((ext_vector_type(8)))  float  v8f;
typedef __attribute__((ext_vector_type(4)))  unsigned int u32x4;
typedef __attribute__((ext_vector_type(8)))  int  i32x8;
typedef __attribute__((ext_vector_type(4)))  int  i32x4;

#if defined(__has_builtin)
# if __has_builtin(__builtin_amdgcn_tensor_load_to_lds) && __has_builtin(__builtin_amdgcn_s_wait_tensorcnt)
#  define HAVE_TDM 1
# endif
#endif
#ifndef HAVE_TDM
# define HAVE_TDM 0
#endif

#define STR2(x) #x
#define STR(x) STR2(x)
#if HAVE_TDM
#pragma message("CDNA5 probe: TDM builtin path ENABLED, clang " STR(__clang_major__))
#else
#pragma message("CDNA5 probe: TDM builtin path DISABLED (fallback staging), clang " STR(__clang_major__))
#endif

union ABFrag { v16bf v; v8bf h[2]; };

__device__ __forceinline__ unsigned short f2bf(float f) {
  unsigned int u = __float_as_uint(f);
  u += 0x7FFFu + ((u >> 16) & 1u);     // round-to-nearest-even
  return (unsigned short)(u >> 16);
}
__device__ __forceinline__ float bf2f(unsigned short h) {
  return __uint_as_float(((unsigned int)h) << 16);
}
__device__ __forceinline__ v8f wmma_bf16(v16bf a, v16bf b, v8f c) {
  return __builtin_amdgcn_wmma_f32_16x16x32_bf16(false, a, false, b, (short)0, c, false, false);
}

// ---------------------------------------------------------------------------
// Elementwise converters
// ---------------------------------------------------------------------------
__global__ __launch_bounds__(256)
void cvt_input(const float* __restrict__ x, const unsigned char* __restrict__ mask,
               unsigned short* __restrict__ act, unsigned char* __restrict__ m8,
               long long total) {
  long long idx = (long long)blockIdx.x * 256 + threadIdx.x;
  if (idx >= total) return;
  long long vox = idx >> 7;
  float v = x[idx] * (mask[vox] ? 1.f : 0.f);
  act[idx] = f2bf(v);
  if ((idx & 127) == 0) m8[vox] = mask[vox] ? 1 : 0;
}

// conv weights (3,3,3,Ci,Co) f32 -> bf16 [tap][Co][Ci]
__global__ __launch_bounds__(256)
void cvt_conv_w(const float* __restrict__ w, unsigned short* __restrict__ wt,
                int Ci, int Co, long long total) {
  long long idx = (long long)blockIdx.x * 256 + threadIdx.x;
  if (idx >= total) return;
  int co = (int)(idx % Co);
  long long t = idx / Co;
  int ci = (int)(t % Ci);
  int tap = (int)(t / Ci);
  wt[((size_t)tap * Co + co) * (size_t)Ci + ci] = f2bf(w[idx]);
}

// FC weights (K,N) f32 -> bf16 [Npad][K] transposed, zero-fill n >= N
__global__ __launch_bounds__(256)
void cvt_fc_w(const float* __restrict__ w, unsigned short* __restrict__ wt,
              int K, int N, long long total) {
  long long idx = (long long)blockIdx.x * 256 + threadIdx.x;
  if (idx >= total) return;
  int k = (int)(idx % K);
  int n = (int)(idx / K);
  float v = (n < N) ? w[(size_t)k * N + n] : 0.f;
  wt[idx] = f2bf(v);
}

// ---------------------------------------------------------------------------
// Masked 2x2x2 max pool + mask propagation
// ---------------------------------------------------------------------------
__global__ __launch_bounds__(256)
void maxpool2_mask(const unsigned short* __restrict__ in, const unsigned char* __restrict__ m_in,
                   unsigned short* __restrict__ out, unsigned char* __restrict__ m_out,
                   int Do, int Co, long long total) {
  long long idx = (long long)blockIdx.x * 256 + threadIdx.x;
  if (idx >= total) return;
  int c  = (int)(idx % Co);
  long long vo = idx / Co;
  int xo = (int)(vo % Do); long long t = vo / Do;
  int yo = (int)(t % Do);  t /= Do;
  int zo = (int)(t % Do);
  int b  = (int)(t / Do);
  int Din = Do * 2;
  float best = -3.4e38f; int any = 0;
  for (int dz = 0; dz < 2; ++dz)
    for (int dy = 0; dy < 2; ++dy)
      for (int dx = 0; dx < 2; ++dx) {
        long long vi = (((long long)b * Din + (2 * zo + dz)) * Din + (2 * yo + dy)) * Din + (2 * xo + dx);
        if (m_in[vi]) { any = 1; best = fmaxf(best, bf2f(in[vi * Co + c])); }
      }
  out[vo * Co + c] = f2bf(any ? best : 0.f);
  if (c == 0) m_out[vo] = (unsigned char)any;
}

// Global max over occupied voxels at res 2 -> g[16][512] (rows 8..15 zero)
__global__ __launch_bounds__(256)
void global_maxpool(const unsigned short* __restrict__ in, const unsigned char* __restrict__ msk,
                    unsigned short* __restrict__ g) {
  int idx = blockIdx.x * 256 + threadIdx.x;
  int c = idx & 511;
  int row = idx >> 9;
  float val = 0.f;
  if (row < 8) {
    float best = -3.4e38f;
    for (int i = 0; i < 8; ++i) {
      int v = row * 8 + i;
      if (msk[v]) best = fmaxf(best, bf2f(in[(size_t)v * 512 + c]));
    }
    val = best;
  }
  g[idx] = f2bf(val);
}

// ---------------------------------------------------------------------------
// Implicit-GEMM 3x3x3 conv + BN + ReLU + mask, bf16 WMMA.
// Block: 256 threads = 8 waves. Tile: 32 voxels x 128 out-channels.
// Flattened K-chunk loop (27 taps x Ci/32), LDS DOUBLE-BUFFERED:
//   one barrier per chunk; weight tile q+1 via TDM (tensor_load_to_lds) and
//   A tile q+1 staged while chunk q runs on the WMMA pipes.
// All index math is shift/mask (D, Ci are powers of two).
// ---------------------------------------------------------------------------
__global__ __launch_bounds__(256)
void conv3_bn_relu_wmma(const unsigned short* __restrict__ in,   // [B*D^3][Ci] bf16 (pre-masked)
                        const unsigned short* __restrict__ wt,   // [27][Co][Ci] bf16
                        const float* __restrict__ bn_g, const float* __restrict__ bn_b,
                        const float* __restrict__ bn_m, const float* __restrict__ bn_v,
                        const unsigned char* __restrict__ msk,
                        unsigned short* __restrict__ out,        // [B*D^3][Co] bf16
                        int dLog, int ciLog, int Co) {
  __shared__ unsigned short Alds[2][32][48];   // 96B rows -> 16B-aligned b128 frag reads
  __shared__ unsigned short Blds[2][128][48];  // stride-48 via TDM pad feature / manual pad

  const int D     = 1 << dLog;
  const int kcLog = ciLog - 5;                 // chunks of 32 ci
  const int kcMsk = (1 << kcLog) - 1;
  const int nq    = 27 << kcLog;

  const int tid  = threadIdx.x;
  const int lane = tid & 31;
  const int wave = tid >> 5;
  const int wm   = wave >> 2;
  const int wn   = wave & 3;
  const int co_base = blockIdx.y * 128;
  const int m_base  = blockIdx.x * 32;

  // A staging mapping: thread -> (voxel row, 4-element ci segment)
  const int ar  = tid >> 3;
  const int seg = tid & 7;
  const int vox = m_base + ar;
  const int ax = vox & (D - 1);
  const int ay = (vox >> dLog) & (D - 1);
  const int az = (vox >> (2 * dLog)) & (D - 1);
  const int ab = vox >> (3 * dLog);

  v8f acc0 = {}; v8f acc1 = {};

  auto stageQ = [&](int q) {
    const int buf = q & 1;
    const int tap = q >> kcLog;
    const int kc  = q & kcMsk;
#if HAVE_TDM
    if (tid == 0) {
      unsigned long long ga =
          (unsigned long long)(const void*)(wt + (((size_t)(tap * Co + co_base)) << ciLog) + kc * 32);
      unsigned int ldsb = (unsigned int)(unsigned long long)(const void*)&Blds[buf][0][0];
      u32x4 g0; i32x8 g1;
      i32x4 g2 = {0, 0, 0, 0}; i32x4 g3 = {0, 0, 0, 0};
      g0[0] = 1u;                                                   // count=1
      g0[1] = ldsb;                                                 // lds_addr
      g0[2] = (unsigned int)ga;                                     // global_addr lo
      g0[3] = (unsigned int)((ga >> 32) & 0x01FFFFFFull) | (2u << 30); // hi | type=2
      // data_size=2B | pad_enable | pad every 16 dwords (32 bf16) | pad 8 dwords (16 bf16)
      g1[0] = (int)((1u << 16) | (1u << 20) | (3u << 22) | (7u << 25));
      g1[1] = (int)((unsigned)(1 << ciLog) << 16);                  // tensor_dim0 lo
      g1[2] = (int)(((unsigned)(1 << ciLog) >> 16) | (((unsigned)(27 * Co) & 0xFFFFu) << 16));
      g1[3] = (int)((((unsigned)(27 * Co)) >> 16) | (32u << 16));   // tile_dim0=32
      g1[4] = 128;                                                  // tile_dim1=128
      g1[5] = (1 << ciLog);                                         // row stride = Ci
      g1[6] = 0; g1[7] = 0;
# if __clang_major__ >= 23
      i32x8 g4 = {0, 0, 0, 0, 0, 0, 0, 0};
      __builtin_amdgcn_tensor_load_to_lds(g0, g1, g2, g3, g4, 0);
# else
      __builtin_amdgcn_tensor_load_to_lds(g0, g1, g2, g3, 0);
# endif
    }
#else
    {
      int bco  = tid >> 1;
      int half = tid & 1;
      const unsigned short* src =
          wt + (((size_t)(tap * Co + co_base + bco)) << ciLog) + kc * 32 + half * 16;
      unsigned short* dst = &Blds[buf][bco][half * 16];
      *(unsigned long long*)(dst + 0)  = *(const unsigned long long*)(src + 0);
      *(unsigned long long*)(dst + 4)  = *(const unsigned long long*)(src + 4);
      *(unsigned long long*)(dst + 8)  = *(const unsigned long long*)(src + 8);
      *(unsigned long long*)(dst + 12) = *(const unsigned long long*)(src + 12);
    }
#endif
    // A tile (zero-fill outside the volume)
    int dz = tap / 9, dy = (tap / 3) % 3, dx = tap % 3;
    int zn = az + dz - 1, yn = ay + dy - 1, xn = ax + dx - 1;
    unsigned long long aval = 0ull;
    if ((unsigned)zn < (unsigned)D && (unsigned)yn < (unsigned)D && (unsigned)xn < (unsigned)D) {
      size_t voxn = ((((((size_t)ab << dLog) | (unsigned)zn) << dLog) | (unsigned)yn) << dLog) | (unsigned)xn;
      aval = *(const unsigned long long*)(in + (voxn << ciLog) + kc * 32 + seg * 4);
    }
    *(unsigned long long*)&Alds[buf][ar][seg * 4] = aval;
  };

  stageQ(0);
  for (int q = 0; q < nq; ++q) {
    const int buf = q & 1;
#if HAVE_TDM
    if (tid == 0) __builtin_amdgcn_s_wait_tensorcnt(0);   // TDM(q) landed
#endif
    __syncthreads();                                      // stage(q) visible, compute(q-1) done
    if (q + 1 < nq) stageQ(q + 1);                        // overlaps with compute(q)

    ABFrag a, b0, b1;
    int arw = (lane & 15) + wm * 16;
    int kh  = (lane >> 4) * 16;
    a.h[0] = *(const v8bf*)&Alds[buf][arw][kh];
    a.h[1] = *(const v8bf*)&Alds[buf][arw][kh + 8];
    int bc0 = wn * 32 + (lane & 15);
    int kb  = (lane >> 4) * 16;
    b0.h[0] = *(const v8bf*)&Blds[buf][bc0][kb];
    b0.h[1] = *(const v8bf*)&Blds[buf][bc0][kb + 8];
    b1.h[0] = *(const v8bf*)&Blds[buf][bc0 + 16][kb];
    b1.h[1] = *(const v8bf*)&Blds[buf][bc0 + 16][kb + 8];
    acc0 = wmma_bf16(a.v, b0.v, acc0);
    acc1 = wmma_bf16(a.v, b1.v, acc1);
  }

  // Epilogue: BN (folded) + ReLU + mask, bf16 store
  int colL = lane & 15;
  int half = lane >> 4;
#pragma unroll
  for (int j = 0; j < 2; ++j) {
    int c = co_base + wn * 32 + j * 16 + colL;
    float sc = bn_g[c] * rsqrtf(bn_v[c] + BN_EPS_F);
    float sh = bn_b[c] - bn_m[c] * sc;
    v8f acc = j ? acc1 : acc0;
#pragma unroll
    for (int r = 0; r < 8; ++r) {
      int orow = r + half * 8;
      size_t ovox = (size_t)m_base + (size_t)wm * 16 + orow;
      float y = acc[r] * sc + sh;
      y = fmaxf(y, 0.f);
      y *= (float)msk[ovox];
      out[ovox * (size_t)Co + c] = f2bf(y);
    }
  }
}

// ---------------------------------------------------------------------------
// FC GEMM: A = g[16][K] bf16 (rows 8..15 zero), Wt = [Npad][K] bf16 transposed.
// ---------------------------------------------------------------------------
__global__ __launch_bounds__(256)
void fc_wmma_k(const unsigned short* __restrict__ A, const unsigned short* __restrict__ Wt,
               const float* __restrict__ bias, int K, int N,
               unsigned short* __restrict__ outg, float* __restrict__ outf, int do_relu) {
  int lane = threadIdx.x & 31;
  int wave = threadIdx.x >> 5;
  int nbase = blockIdx.x * 128 + wave * 16;
  int rl = lane & 15, half = lane >> 4;
  v8f acc = {};
  const unsigned short* arow = A + (size_t)rl * K;
  const unsigned short* brow = Wt + (size_t)(nbase + rl) * K;
  for (int k = 0; k < K; k += 32) {
    ABFrag a, b;
    a.h[0] = *(const v8bf*)(arow + k + half * 16);
    a.h[1] = *(const v8bf*)(arow + k + half * 16 + 8);
    b.h[0] = *(const v8bf*)(brow + k + half * 16);
    b.h[1] = *(const v8bf*)(brow + k + half * 16 + 8);
    acc = wmma_bf16(a.v, b.v, acc);
  }
  int c = nbase + rl;
  float bv = (c < N) ? bias[c] : 0.f;
#pragma unroll
  for (int r = 0; r < 8; ++r) {
    int orow = r + half * 8;
    float y = acc[r] + bv;
    if (do_relu) y = fmaxf(y, 0.f);
    if (orow >= 8) y = 0.f;
    if (outf) {
      if (orow < 8 && c < N) outf[(size_t)orow * N + c] = y;
    } else {
      outg[(size_t)orow * N + c] = f2bf(y);
    }
  }
}

// ---------------------------------------------------------------------------
// Host orchestration
// ---------------------------------------------------------------------------
static inline int ilog2i(int v) { int l = 0; while ((1 << l) < v) ++l; return l; }

extern "C" void kernel_launch(void* const* d_in, const int* in_sizes, int n_in,
                              void* d_out, int out_size, void* d_ws, size_t ws_size,
                              hipStream_t stream) {
  (void)in_sizes; (void)n_in; (void)out_size; (void)ws_size;
  const float* x = (const float*)d_in[0];
  const unsigned char* mask = (const unsigned char*)d_in[1];
  static const int CI[10] = {128, 128, 256, 256, 256, 512, 512, 512, 512, 512};
  static const int CO[10] = {128, 256, 256, 256, 512, 512, 512, 512, 512, 512};
  static const int NCONV[4] = {1, 3, 3, 3};

  char* ws = (char*)d_ws;
  size_t off = 0;
  auto alloc = [&](size_t bytes) -> void* {
    void* p = ws + off;
    off += (bytes + 255) & ~(size_t)255;
    return p;
  };
  unsigned short* ACT0 = (unsigned short*)alloc((size_t)8 * 32 * 32 * 32 * 128 * 2);
  unsigned short* ACT1 = (unsigned short*)alloc((size_t)8 * 32 * 32 * 32 * 128 * 2);
  unsigned short* WT   = (unsigned short*)alloc((size_t)4096 * 4096 * 2);
  unsigned char* MB[5];
  const int Ds[5] = {32, 16, 8, 4, 2};
  for (int i = 0; i < 5; ++i) MB[i] = (unsigned char*)alloc((size_t)8 * Ds[i] * Ds[i] * Ds[i]);
  unsigned short* G0 = (unsigned short*)alloc((size_t)16 * 512 * 2);
  unsigned short* G1 = (unsigned short*)alloc((size_t)16 * 4096 * 2);
  unsigned short* G2 = (unsigned short*)alloc((size_t)16 * 4096 * 2);

  long long totalIn = 8LL * 32 * 32 * 32 * 128;
  cvt_input<<<(unsigned)((totalIn + 255) / 256), 256, 0, stream>>>(x, mask, ACT0, MB[0], totalIn);

  unsigned short* bufs[2] = {ACT0, ACT1};
  int cur = 0, layer = 0, D = 32;
  for (int blk = 0; blk < 4; ++blk) {
    for (int j = 0; j < NCONV[blk]; ++j) {
      int Ci = CI[layer], Co = CO[layer];
      long long tw = 27LL * Ci * Co;
      cvt_conv_w<<<(unsigned)((tw + 255) / 256), 256, 0, stream>>>(
          (const float*)d_in[2 + layer], WT, Ci, Co, tw);
      long long M = 8LL * D * D * D;
      dim3 g((unsigned)(M / 32), (unsigned)(Co / 128));
      conv3_bn_relu_wmma<<<g, 256, 0, stream>>>(
          bufs[cur], WT,
          (const float*)d_in[12 + 4 * layer + 0], (const float*)d_in[12 + 4 * layer + 1],
          (const float*)d_in[12 + 4 * layer + 2], (const float*)d_in[12 + 4 * layer + 3],
          MB[blk], bufs[cur ^ 1], ilog2i(D), ilog2i(Ci), Co);
      cur ^= 1; ++layer;
    }
    int Co = CO[layer - 1];
    int Do = D / 2;
    long long tp = 8LL * Do * Do * Do * Co;
    maxpool2_mask<<<(unsigned)((tp + 255) / 256), 256, 0, stream>>>(
        bufs[cur], MB[blk], bufs[cur ^ 1], MB[blk + 1], Do, Co, tp);
    cur ^= 1; D = Do;
  }

  global_maxpool<<<32, 256, 0, stream>>>(bufs[cur], MB[4], G0);

  long long t1 = 4096LL * 512;
  cvt_fc_w<<<(unsigned)((t1 + 255) / 256), 256, 0, stream>>>((const float*)d_in[52], WT, 512, 4096, t1);
  fc_wmma_k<<<32, 256, 0, stream>>>(G0, WT, (const float*)d_in[55], 512, 4096, G1, (float*)nullptr, 1);

  long long t2 = 4096LL * 4096;
  cvt_fc_w<<<(unsigned)((t2 + 255) / 256), 256, 0, stream>>>((const float*)d_in[53], WT, 4096, 4096, t2);
  fc_wmma_k<<<32, 256, 0, stream>>>(G1, WT, (const float*)d_in[56], 4096, 4096, G2, (float*)nullptr, 1);

  long long t3 = 128LL * 4096;   // N=40 padded to 128 (zero-filled)
  cvt_fc_w<<<(unsigned)((t3 + 255) / 256), 256, 0, stream>>>((const float*)d_in[54], WT, 4096, 40, t3);
  fc_wmma_k<<<1, 256, 0, stream>>>(G2, WT, (const float*)d_in[57], 4096, 40,
                                   (unsigned short*)nullptr, (float*)d_out, 0);
}
